// VectorQuantizer_65171833749596
// MI455X (gfx1250) — matrix operand order repaired and checked by reading, not compiled, not run on hardware
//
#include <hip/hip_runtime.h>

// ---- problem constants (match reference) ----
#define K_EMB   4096
#define D_DIM   128
#define B_ROWS  32768

typedef __attribute__((ext_vector_type(16))) __bf16 v16bf;
typedef __attribute__((ext_vector_type(8)))  __bf16 v8bf;
typedef __attribute__((ext_vector_type(8)))  float  v8f;

// ---------------------------------------------------------------------------
// 1) Split fp32 -> bf16 hi + bf16 lo residual (Markidis 3-product split)
// ---------------------------------------------------------------------------
__global__ __launch_bounds__(256)
void vq_split_bf16(const float* __restrict__ src, __bf16* __restrict__ hi,
                   __bf16* __restrict__ lo, int n) {
    int i = blockIdx.x * 256 + threadIdx.x;
    if (i < n) {
        float x = src[i];
        __bf16 h = (__bf16)x;
        hi[i] = h;
        lo[i] = (__bf16)(x - (float)h);
    }
}

// ---------------------------------------------------------------------------
// 2) Per-codebook-row squared norms  e2[k] = sum_d E[k][d]^2   (fp32 exact)
// ---------------------------------------------------------------------------
__global__ __launch_bounds__(256)
void vq_rownorm(const float* __restrict__ e, float* __restrict__ e2) {
    int r = blockIdx.x * 256 + threadIdx.x;
    if (r < K_EMB) {
        const float4* p = (const float4*)(e + (size_t)r * D_DIM);
        float s = 0.f;
#pragma unroll 8
        for (int i = 0; i < D_DIM / 4; ++i) {
            float4 v = p[i];
            s += v.x * v.x + v.y * v.y + v.z * v.z + v.w * v.w;
        }
        e2[r] = s;
    }
}

__global__ __launch_bounds__(256)
void vq_zero(float* __restrict__ p, int n) {
    int i = blockIdx.x * 256 + threadIdx.x;
    if (i < n) p[i] = 0.f;
}

// ---------------------------------------------------------------------------
// 3) Main fused GEMM + argmin kernel.
//    dist[m][n] = e2[n] - 2 * dot(X[m], E[n])   (||x||^2 constant per row)
//    dot via split-bf16 WMMA: hi*hi + hi*lo + lo*hi  (~fp32 accurate)
//    Block = 8 waves, each wave owns 16 rows of X; E tiles staged via LDS.
// ---------------------------------------------------------------------------
__global__ __launch_bounds__(256)
void vq_argmin(const __bf16* __restrict__ xhi, const __bf16* __restrict__ xlo,
               const __bf16* __restrict__ ehi, const __bf16* __restrict__ elo,
               const float* __restrict__ e2,
               int* __restrict__ ws_idx, float* __restrict__ out_idx) {
    __shared__ __align__(32) __bf16 lds_hi[16 * D_DIM];  // 4 KB
    __shared__ __align__(32) __bf16 lds_lo[16 * D_DIM];  // 4 KB

    const int t    = threadIdx.x;
    const int lane = t & 31;           // wave32
    const int wave = t >> 5;
    const int m0   = blockIdx.x * 128 + wave * 16;

    // A fragment (16x32 bf16, ISA 7.12.2): lane L<16 -> row M=L, K {0..7,16..23}
    //                                      lane L>=16 -> row M=L-16, K {8..15,24..31}
    const int arow = m0 + (lane & 15);
    const int aoff = (lane >= 16) ? 8 : 0;

    v16bf Ah[4], Al[4];   // 4 chunks of K=32 cover D=128; hi + lo residual
#pragma unroll
    for (int c = 0; c < 4; ++c) {
        const size_t base = (size_t)arow * D_DIM + c * 32 + aoff;
        v8bf h0 = *(const v8bf*)(xhi + base);
        v8bf h1 = *(const v8bf*)(xhi + base + 16);
        Ah[c] = __builtin_shufflevector(h0, h1, 0,1,2,3,4,5,6,7,8,9,10,11,12,13,14,15);
        v8bf l0 = *(const v8bf*)(xlo + base);
        v8bf l1 = *(const v8bf*)(xlo + base + 16);
        Al[c] = __builtin_shufflevector(l0, l1, 0,1,2,3,4,5,6,7,8,9,10,11,12,13,14,15);
    }

    // C layout: lanes 0-15 VGPR r -> M=r ; lanes 16-31 VGPR r -> M=8+r ; N=lane&15
    float bestV[8];
    int   bestI[8];
#pragma unroll
    for (int r = 0; r < 8; ++r) { bestV[r] = 3.402823466e38f; bestI[r] = 0; }

    const int ldr  = t >> 4;          // cooperative staging: row 0..15
    const int ldc  = (t & 15) * 8;    // 8 bf16 (16B) per thread
    // B fragment (32x16 bf16): lane l<16 -> col N=l, K {0..7,16..23};
    //                          lane l>=16 -> col N=l-16, K {8..15,24..31}
    const int brow = lane & 15;
    const int boff = (lane >= 16) ? 8 : 0;

    for (int nt = 0; nt < K_EMB / 16; ++nt) {
        const int n0 = nt * 16;
        __syncthreads();
        *(v8bf*)(lds_hi + ldr * D_DIM + ldc) =
            *(const v8bf*)(ehi + (size_t)(n0 + ldr) * D_DIM + ldc);
        *(v8bf*)(lds_lo + ldr * D_DIM + ldc) =
            *(const v8bf*)(elo + (size_t)(n0 + ldr) * D_DIM + ldc);
        if (nt + 1 < K_EMB / 16) {   // global_prefetch_b8 for next tile
            __builtin_prefetch(ehi + (size_t)(n0 + 16 + ldr) * D_DIM + ldc, 0, 1);
            __builtin_prefetch(elo + (size_t)(n0 + 16 + ldr) * D_DIM + ldc, 0, 1);
        }
        __syncthreads();

        v8f acc = {0.f, 0.f, 0.f, 0.f, 0.f, 0.f, 0.f, 0.f};
#pragma unroll
        for (int c = 0; c < 4; ++c) {
            const int kb = c * 32 + boff;
            v16bf Bh = __builtin_shufflevector(*(const v8bf*)(lds_hi + brow * D_DIM + kb), *(const v8bf*)(lds_hi + brow * D_DIM + kb + 16), 0,1,2,3,4,5,6,7,8,9,10,11,12,13,14,15);
            v16bf Bl = __builtin_shufflevector(*(const v8bf*)(lds_lo + brow * D_DIM + kb), *(const v8bf*)(lds_lo + brow * D_DIM + kb + 16), 0,1,2,3,4,5,6,7,8,9,10,11,12,13,14,15);
            // dot ~= hi*hi + hi*lo + lo*hi  (accumulator D->C chaining: no hazard)
            acc = __builtin_amdgcn_wmma_f32_16x16x32_bf16(false, Ah[c], false, Bh,
                                                          (short)0, acc, false, false);
            acc = __builtin_amdgcn_wmma_f32_16x16x32_bf16(false, Ah[c], false, Bl,
                                                          (short)0, acc, false, false);
            acc = __builtin_amdgcn_wmma_f32_16x16x32_bf16(false, Al[c], false, Bh,
                                                          (short)0, acc, false, false);
        }

        const int   n   = n0 + (lane & 15);
        const float e2n = e2[n];
#pragma unroll
        for (int r = 0; r < 8; ++r) {
            float d = __builtin_fmaf(-2.0f, acc[r], e2n);
            // strict '<' with ascending n keeps the FIRST minimum (jnp.argmin)
            if (d < bestV[r]) { bestV[r] = d; bestI[r] = n; }
        }
    }

    // Reduce the 16 column-slots of each row across each 16-lane half.
#pragma unroll
    for (int r = 0; r < 8; ++r) {
#pragma unroll
        for (int m = 1; m < 16; m <<= 1) {
            float ov = __shfl_xor(bestV[r], m, 32);
            int   oi = __shfl_xor(bestI[r], m, 32);
            if (ov < bestV[r] || (ov == bestV[r] && oi < bestI[r])) {
                bestV[r] = ov; bestI[r] = oi;
            }
        }
    }
    if ((lane & 15) == 0) {
        const int rbase = m0 + (lane >> 4) * 8;   // lane0 -> rows m0..m0+7, lane16 -> +8
#pragma unroll
        for (int r = 0; r < 8; ++r) {
            ws_idx[rbase + r]  = bestI[r];
            out_idx[rbase + r] = (float)bestI[r];
        }
    }
}

// ---------------------------------------------------------------------------
// 4) Gather quantized rows (exact fp32) + block-partial sum of (q-x)^2
// ---------------------------------------------------------------------------
__global__ __launch_bounds__(256)
void vq_gather_loss(const float* __restrict__ x, const float* __restrict__ emb,
                    const int* __restrict__ idx, float* __restrict__ outq,
                    float* __restrict__ partial) {
    __shared__ float red[256];
    int i   = blockIdx.x * 256 + threadIdx.x;
    int row = i >> 7;        // / D_DIM
    int col = i & (D_DIM - 1);
    float q  = emb[(size_t)idx[row] * D_DIM + col];
    float xv = x[i];
    outq[i] = q;             // quantized_st == quantized numerically
    float d = q - xv;
    red[threadIdx.x] = d * d;
    __syncthreads();
    for (int s = 128; s > 0; s >>= 1) {
        if (threadIdx.x < s) red[threadIdx.x] += red[threadIdx.x + s];
        __syncthreads();
    }
    if (threadIdx.x == 0) partial[blockIdx.x] = red[0];
}

__global__ __launch_bounds__(256)
void vq_count(const int* __restrict__ idx, float* __restrict__ counts) {
    int r = blockIdx.x * 256 + threadIdx.x;
    if (r < B_ROWS) atomicAdd(&counts[idx[r]], 1.0f);
}

// ---------------------------------------------------------------------------
// 5) Finalize: losses + perplexity (single workgroup)
// ---------------------------------------------------------------------------
__global__ __launch_bounds__(256)
void vq_finalize(const float* __restrict__ partial, int nPartial,
                 const float* __restrict__ counts, float* __restrict__ outs) {
    __shared__ float red[256];
    float s = 0.f;
    for (int i = threadIdx.x; i < nPartial; i += 256) s += partial[i];
    red[threadIdx.x] = s;
    __syncthreads();
    for (int st = 128; st > 0; st >>= 1) {
        if (threadIdx.x < st) red[threadIdx.x] += red[threadIdx.x + st];
        __syncthreads();
    }
    float total = red[0];
    __syncthreads();

    float h = 0.f;
    for (int i = threadIdx.x; i < K_EMB; i += 256) {
        float p = counts[i] * (1.0f / (float)B_ROWS);
        h += p * logf(p + 1e-10f);
    }
    red[threadIdx.x] = h;
    __syncthreads();
    for (int st = 128; st > 0; st >>= 1) {
        if (threadIdx.x < st) red[threadIdx.x] += red[threadIdx.x + st];
        __syncthreads();
    }
    if (threadIdx.x == 0) {
        float e = total / (float)((size_t)B_ROWS * D_DIM);
        outs[0] = e + 0.25f * e;  // vq_loss = q_latent + COMMIT * e_latent
        outs[1] = e;              // e_latent_loss
        outs[2] = e;              // q_latent_loss (numerically identical)
        outs[3] = expf(-red[0]);  // perplexity
    }
}

// ---------------------------------------------------------------------------
// Launcher.  ws layout (bytes):
//   [0)        ws_idx   int[32768]               128 KB
//   [131072)   counts   float[4096]               16 KB   (zeroed each call)
//   [147456)   partial  float[16384]              64 KB   (zeroed each call)
//   [212992)   e2       float[4096]               16 KB
//   [229376)   xhi/xlo/ehi/elo bf16             ~18.2 MB
// ---------------------------------------------------------------------------
extern "C" void kernel_launch(void* const* d_in, const int* in_sizes, int n_in,
                              void* d_out, int out_size, void* d_ws, size_t ws_size,
                              hipStream_t stream) {
    (void)in_sizes; (void)n_in; (void)out_size; (void)ws_size;
    const float* x   = (const float*)d_in[0];   // [B, D] fp32
    const float* emb = (const float*)d_in[1];   // [K, D] fp32
    float* out = (float*)d_out;
    char*  ws  = (char*)d_ws;

    int*    ws_idx  = (int*)(ws + 0);
    float*  counts  = (float*)(ws + 131072);
    float*  partial = (float*)(ws + 147456);
    float*  e2      = (float*)(ws + 212992);
    __bf16* xhi = (__bf16*)(ws + 229376);
    __bf16* xlo = xhi + (size_t)B_ROWS * D_DIM;
    __bf16* ehi = xlo + (size_t)B_ROWS * D_DIM;
    __bf16* elo = ehi + (size_t)K_EMB * D_DIM;

    float* outq   = out;                              // [B*D]
    float* outidx = out + (size_t)B_ROWS * D_DIM;     // [B] indices as float
    float* outs   = outidx + B_ROWS;                  // 4 scalars

    const int nX = B_ROWS * D_DIM;   // 4,194,304
    const int nE = K_EMB * D_DIM;    //   524,288
    const int nGather = nX / 256;    //    16,384 partial blocks

    vq_zero<<<(4096 + 16384 + 255) / 256, 256, 0, stream>>>(counts, 4096 + 16384);
    vq_split_bf16<<<nX / 256, 256, 0, stream>>>(x, xhi, xlo, nX);
    vq_split_bf16<<<nE / 256, 256, 0, stream>>>(emb, ehi, elo, nE);
    vq_rownorm<<<K_EMB / 256, 256, 0, stream>>>(emb, e2);

    vq_argmin<<<B_ROWS / 128, 256, 0, stream>>>(xhi, xlo, ehi, elo, e2, ws_idx, outidx);

    vq_gather_loss<<<nGather, 256, 0, stream>>>(x, emb, ws_idx, outq, partial);
    vq_count<<<B_ROWS / 256, 256, 0, stream>>>(ws_idx, counts);
    vq_finalize<<<1, 256, 0, stream>>>(partial, nGather, counts, outs);
}
